// MsingiBlock_90237262889055
// MI455X (gfx1250) — matrix operand order repaired and checked by reading, not compiled, not run on hardware
//
#include <hip/hip_runtime.h>
#include <math.h>

// ---------------- problem constants ----------------
#define B_   4
#define S_   2048
#define H_   1024
#define NH_  16
#define HD_  64
#define FF_  4096
#define NT_  (B_ * S_)          // 8192 tokens
#define EPS_ 1e-5f

typedef _Float16 v16h __attribute__((ext_vector_type(16)));
typedef _Float16 v8h  __attribute__((ext_vector_type(8)));
typedef float    v8f  __attribute__((ext_vector_type(8)));

// ---------------- WMMA fragment helpers (CDNA5 wave32 layouts) ----------------
__device__ __forceinline__ v16h join16(v8h lo, v8h hi) {
  union { v16h v; v8h h[2]; } u; u.h[0] = lo; u.h[1] = hi; return u.v;
}
// A 16x32 f16: lane(m=lane%16, hf=lane/16): k = hf*8 + {0..7} and 16 + hf*8 + {0..7}
__device__ __forceinline__ v16h frag_a16(const _Float16* row, int hf) {
  return join16(*(const v8h*)(row + hf * 8), *(const v8h*)(row + hf * 8 + 16));
}
// B 32x16 f16 (row = column n of B, contiguous over k): k = hf*8 + {0..7} and 16 + hf*8 + {0..7}
__device__ __forceinline__ v16h frag_b16(const _Float16* row, int hf) {
  return join16(*(const v8h*)(row + hf * 8), *(const v8h*)(row + hf * 8 + 16));
}
__device__ __forceinline__ v8f wmma_f16(v16h a, v16h b, v8f c) {
  return __builtin_amdgcn_wmma_f32_16x16x32_f16(false, a, false, b, (short)0, c,
                                                false, false);
}

// ---------------- CDNA5 async global -> LDS copy (16 bytes / lane) ----------------
// vdst holds the LDS byte offset; generic LDS pointer truncated to 32 bits gives
// the per-workgroup LDS byte address (aperture low bits).
__device__ __forceinline__ void async_cp16(unsigned int lds_off,
                                           const void* gaddr) {
  asm volatile("global_load_async_to_lds_b128 %0, %1, off"
               :: "v"(lds_off), "v"((unsigned long long)(uintptr_t)gaddr)
               : "memory");
}
__device__ __forceinline__ void wait_async0() {
  asm volatile("s_wait_asynccnt 0" ::: "memory");
}
__device__ __forceinline__ unsigned int lds_addr(const void* p) {
  return (unsigned int)(uintptr_t)p;
}

// ---------------- elementwise convert f32 -> f16 ----------------
__global__ __launch_bounds__(256) void cvt_f16_kernel(const float* __restrict__ in,
                                                      _Float16* __restrict__ out) {
  const int i = (blockIdx.x * 256 + threadIdx.x) * 4;
#pragma unroll
  for (int j = 0; j < 4; ++j) out[i + j] = (_Float16)in[i + j];
}

// ---------------- convert + transpose: W[K][N] f32 -> Wt[N][K] f16 (N = 2^lgN) ----
__global__ __launch_bounds__(256) void cvt_tr_kernel(const float* __restrict__ W,
                                                     _Float16* __restrict__ Wt,
                                                     int K, int lgN) {
  const int idx = blockIdx.x * 256 + threadIdx.x;   // over K*N, coalesced read
  const int k = idx >> lgN, n = idx & ((1 << lgN) - 1);
  Wt[(size_t)n * K + k] = (_Float16)W[idx];
}

// ---------------- WMMA GEMM: C[M,N] = A[M,K](f16) @ Bt[N,K](f16) + bias ----------
// block tile 128(M) x 128(N), K-step 32; 8 waves, each wave a 32(M) x 64(N) region.
// Double-buffered LDS filled by async global->LDS copies; one barrier per K-step.
// act: 0 = none, 1 = exact GELU.
__global__ __launch_bounds__(256)
void gemm_wmma_kernel(const _Float16* __restrict__ A, const _Float16* __restrict__ Bt,
                      const float* __restrict__ bias, float* __restrict__ Cf,
                      _Float16* __restrict__ Ch, int M, int N, int K, int act) {
  __shared__ _Float16 sA[2][128 * 40];  // 40-half rows: 80B (16B aligned)
  __shared__ _Float16 sB[2][128 * 40];

  const int tid = threadIdx.x;
  const int wave = tid >> 5, lane = tid & 31, lm = lane & 15, hf = lane >> 4;
  const int wm = wave & 3, wn = wave >> 2;          // 4 x 2 wave grid
  const int m0 = blockIdx.y * 128, n0 = blockIdx.x * 128;

  v8f acc[2][4] = {};

  // staging: 128 rows x 32 halves each for A and B -> 16 halves/thread/tile
  const int srow = tid >> 1, sch = tid & 1;
  const _Float16* agb = A + (size_t)(m0 + srow) * K + sch * 16;
  const _Float16* bgb = Bt + (size_t)(n0 + srow) * K + sch * 16;
  unsigned int aoff[2], boff[2];
#pragma unroll
  for (int p = 0; p < 2; ++p) {
    aoff[p] = lds_addr(&sA[p][srow * 40 + sch * 16]);
    boff[p] = lds_addr(&sB[p][srow * 40 + sch * 16]);
  }
  auto issue = [&](int p, int kt) {
    async_cp16(aoff[p], agb + kt);
    async_cp16(aoff[p] + 16, agb + kt + 8);
    async_cp16(boff[p], bgb + kt);
    async_cp16(boff[p] + 16, bgb + kt + 8);
  };

  issue(0, 0);
  for (int kt = 0; kt < K; kt += 32) {
    const int p = (kt >> 5) & 1;
    wait_async0();           // this wave's copies into buffer p have landed
    __syncthreads();         // everyone's copies landed; prev reads of p^1 done
    if (kt + 32 < K) issue(p ^ 1, kt + 32);
    if (kt + 64 < K) {       // warm L2 two tiles ahead
      __builtin_prefetch(agb + kt + 64, 0, 2);
      __builtin_prefetch(bgb + kt + 64, 0, 2);
    }

    v16h af[2], bf[4];
#pragma unroll
    for (int mt = 0; mt < 2; ++mt)
      af[mt] = frag_a16(&sA[p][(wm * 32 + mt * 16 + lm) * 40], hf);
#pragma unroll
    for (int nt = 0; nt < 4; ++nt)
      bf[nt] = frag_b16(&sB[p][(wn * 64 + nt * 16 + lm) * 40], hf);
#pragma unroll
    for (int mt = 0; mt < 2; ++mt)
#pragma unroll
      for (int nt = 0; nt < 4; ++nt)
        acc[mt][nt] = wmma_f16(af[mt], bf[nt], acc[mt][nt]);
  }

#pragma unroll
  for (int mt = 0; mt < 2; ++mt) {
#pragma unroll
    for (int nt = 0; nt < 4; ++nt) {
      const int n = n0 + wn * 64 + nt * 16 + lm;
      const float bn = bias[n];
#pragma unroll
      for (int r = 0; r < 8; ++r) {
        const int m = m0 + wm * 32 + mt * 16 + hf * 8 + r;  // C layout: m=r+8*hf
        float vv = acc[mt][nt][r] + bn;
        if (act == 1) vv = 0.5f * vv * (1.0f + erff(vv * 0.70710678118654752f));
        if (Cf) Cf[(size_t)m * N + n] = vv;
        else    Ch[(size_t)m * N + n] = (_Float16)vv;
      }
    }
  }
}

// ---------------- flash attention: one (b, h, 128 q-rows) per block ----------------
__global__ __launch_bounds__(256)
void flash_attn_kernel(const _Float16* __restrict__ q, const _Float16* __restrict__ k,
                       const _Float16* __restrict__ v, float* __restrict__ out) {
  __shared__ _Float16 sK[32 * 72];        // 32 key rows x 64 (pad 72 -> 144B rows)
  __shared__ _Float16 sVt[64 * 40];       // V transposed: [n=64][kdim=32] pad 40
  __shared__ _Float16 sP[8 * 16 * 40];    // per-wave P tile 16x32 pad 40

  const int tid = threadIdx.x;
  const int wave = tid >> 5, lane = tid & 31, lm = lane & 15, hf = lane >> 4;
  const int qblk = blockIdx.x, hh = blockIdx.y, bz = blockIdx.z;
  const size_t bb = (size_t)bz * S_ * H_;
  const int qr = qblk * 128 + wave * 16 + lm;
  const _Float16* qrow = q + bb + (size_t)qr * H_ + hh * HD_;

  v16h aq[2];
#pragma unroll
  for (int c = 0; c < 2; ++c) aq[c] = frag_a16(qrow + c * 32, hf);

  float mrow[8], lrow[8];
#pragma unroll
  for (int r = 0; r < 8; ++r) { mrow[r] = -1e30f; lrow[r] = 0.0f; }
  v8f o[4] = {};

  _Float16* pw = &sP[wave * (16 * 40)];
  const int krow = tid >> 3, kch = tid & 7;       // K stage (async, 16B/lane)
  const int vn = tid & 63, vk0 = (tid >> 6) * 8;  // V transpose stage (scalar)
  const unsigned int skoff = lds_addr(&sK[krow * 72 + kch * 8]);
  const _Float16* kgb = k + bb + (size_t)krow * H_ + hh * HD_ + kch * 8;

  for (int j = 0; j < S_; j += 32) {
    async_cp16(skoff, kgb + (size_t)j * H_);
#pragma unroll
    for (int i = 0; i < 8; ++i)
      sVt[vn * 40 + vk0 + i] = v[bb + (size_t)(j + vk0 + i) * H_ + hh * HD_ + vn];
    wait_async0();
    __syncthreads();

    // scores: 16 x 32 as two 16x16 tiles; K-dim = HD = 64 (two WMMA each)
    v8f s[2] = {};
#pragma unroll
    for (int nt = 0; nt < 2; ++nt) {
      const _Float16* kr = &sK[(nt * 16 + lm) * 72];
      s[nt] = wmma_f16(aq[0], frag_b16(kr, hf), s[nt]);
      s[nt] = wmma_f16(aq[1], frag_b16(kr + 32, hf), s[nt]);
    }

    // online softmax (rows live across the 16 lanes of each half)
    float rmax[8];
#pragma unroll
    for (int r = 0; r < 8; ++r) rmax[r] = fmaxf(s[0][r], s[1][r]) * 0.125f;
#pragma unroll
    for (int off = 1; off < 16; off <<= 1)
#pragma unroll
      for (int r = 0; r < 8; ++r)
        rmax[r] = fmaxf(rmax[r], __shfl_xor(rmax[r], off, 32));

    float alpha[8], rsum[8];
#pragma unroll
    for (int r = 0; r < 8; ++r) {
      const float mnew = fmaxf(mrow[r], rmax[r]);
      alpha[r] = expf(mrow[r] - mnew);
      mrow[r] = mnew;
      s[0][r] = expf(s[0][r] * 0.125f - mnew);
      s[1][r] = expf(s[1][r] * 0.125f - mnew);
      rsum[r] = s[0][r] + s[1][r];
    }
#pragma unroll
    for (int off = 1; off < 16; off <<= 1)
#pragma unroll
      for (int r = 0; r < 8; ++r) rsum[r] += __shfl_xor(rsum[r], off, 32);
#pragma unroll
    for (int r = 0; r < 8; ++r) lrow[r] = lrow[r] * alpha[r] + rsum[r];
#pragma unroll
    for (int t = 0; t < 4; ++t)
#pragma unroll
      for (int r = 0; r < 8; ++r) o[t][r] *= alpha[r];

    // spill P (C layout) to per-wave LDS, re-read as A fragments
#pragma unroll
    for (int r = 0; r < 8; ++r) {
      pw[(hf * 8 + r) * 40 + lm]      = (_Float16)s[0][r];
      pw[(hf * 8 + r) * 40 + 16 + lm] = (_Float16)s[1][r];
    }
    asm volatile("s_wait_dscnt 0" ::: "memory");  // LDS in-order within wave

    const v16h ap = frag_a16(pw + lm * 40, hf);
#pragma unroll
    for (int t = 0; t < 4; ++t)
      o[t] = wmma_f16(ap, frag_b16(&sVt[(t * 16 + lm) * 40], hf), o[t]);

    __syncthreads();
  }

#pragma unroll
  for (int r = 0; r < 8; ++r) {
    const float inv = 1.0f / lrow[r];
    const int row = qblk * 128 + wave * 16 + hf * 8 + r;
    const size_t ob = bb + (size_t)row * H_ + hh * HD_ + lm;
#pragma unroll
    for (int t = 0; t < 4; ++t) out[ob + t * 16] = o[t][r] * inv;
  }
}

// ---------------- fused residual add + LayerNorm (one 1024-wide row / block) ------
__global__ __launch_bounds__(256)
void add_ln_kernel(const float* __restrict__ a, const float* __restrict__ res,
                   const float* __restrict__ gamma, const float* __restrict__ beta,
                   float* __restrict__ outf, _Float16* __restrict__ outh) {
  __shared__ float s1[256], s2[256];
  const int tid = threadIdx.x;
  const size_t base = (size_t)blockIdx.x * H_;
  float vv[4], sum = 0.0f, sq = 0.0f;
#pragma unroll
  for (int i = 0; i < 4; ++i) {
    const int c = tid + i * 256;
    const float t = a[base + c] + res[base + c];
    vv[i] = t; sum += t; sq += t * t;
  }
  s1[tid] = sum; s2[tid] = sq;
  __syncthreads();
  for (int s = 128; s > 0; s >>= 1) {
    if (tid < s) { s1[tid] += s1[tid + s]; s2[tid] += s2[tid + s]; }
    __syncthreads();
  }
  const float mean = s1[0] * (1.0f / (float)H_);
  const float var = s2[0] * (1.0f / (float)H_) - mean * mean;
  const float rstd = rsqrtf(var + EPS_);
#pragma unroll
  for (int i = 0; i < 4; ++i) {
    const int c = tid + i * 256;
    const float o = (vv[i] - mean) * rstd * gamma[c] + beta[c];
    outf[base + c] = o;
    if (outh) outh[base + c] = (_Float16)o;
  }
}

// ---------------- host launcher ----------------
extern "C" void kernel_launch(void* const* d_in, const int* in_sizes, int n_in,
                              void* d_out, int out_size, void* d_ws, size_t ws_size,
                              hipStream_t stream) {
  const float* x  = (const float*)d_in[0];
  const float* Wq = (const float*)d_in[1];  const float* bq = (const float*)d_in[2];
  const float* Wk = (const float*)d_in[3];  const float* bk = (const float*)d_in[4];
  const float* Wv = (const float*)d_in[5];  const float* bv = (const float*)d_in[6];
  const float* Wi = (const float*)d_in[7];  const float* bi = (const float*)d_in[8];
  const float* Wo = (const float*)d_in[9];  const float* bo = (const float*)d_in[10];
  const float* g1 = (const float*)d_in[11]; const float* b1 = (const float*)d_in[12];
  const float* g2 = (const float*)d_in[13]; const float* b2 = (const float*)d_in[14];
  float* out = (float*)d_out;

  char* ws = (char*)d_ws;
  size_t o = 0;
  auto carve = [&](size_t bytes) -> void* {
    void* p = ws + o;
    o += (bytes + 255) & ~(size_t)255;
    return p;
  };
  _Float16* xf16 = (_Float16*)carve((size_t)NT_ * H_ * 2);
  _Float16* WqT  = (_Float16*)carve((size_t)H_ * H_ * 2);
  _Float16* WkT  = (_Float16*)carve((size_t)H_ * H_ * 2);
  _Float16* WvT  = (_Float16*)carve((size_t)H_ * H_ * 2);
  _Float16* WiT  = (_Float16*)carve((size_t)H_ * FF_ * 2);
  _Float16* WoT  = (_Float16*)carve((size_t)H_ * FF_ * 2);
  _Float16* qf   = (_Float16*)carve((size_t)NT_ * H_ * 2);
  _Float16* kf   = (_Float16*)carve((size_t)NT_ * H_ * 2);
  _Float16* vf   = (_Float16*)carve((size_t)NT_ * H_ * 2);
  float*    attn = (float*)   carve((size_t)NT_ * H_ * 4);
  float*    h1   = (float*)   carve((size_t)NT_ * H_ * 4);
  _Float16* h1h  = (_Float16*)carve((size_t)NT_ * H_ * 2);
  _Float16* f1   = (_Float16*)carve((size_t)NT_ * FF_ * 2);
  float*    f2   = (float*)   carve((size_t)NT_ * H_ * 4);
  if (o > ws_size) return;   // constant per run: deterministic

  // 1) convert activations and weights to f16 (weights transposed to [N][K])
  cvt_f16_kernel<<<(NT_ * H_) / 1024, 256, 0, stream>>>(x, xf16);
  cvt_tr_kernel<<<(H_ * H_) / 256, 256, 0, stream>>>(Wq, WqT, H_, 10);
  cvt_tr_kernel<<<(H_ * H_) / 256, 256, 0, stream>>>(Wk, WkT, H_, 10);
  cvt_tr_kernel<<<(H_ * H_) / 256, 256, 0, stream>>>(Wv, WvT, H_, 10);
  cvt_tr_kernel<<<(H_ * FF_) / 256, 256, 0, stream>>>(Wi, WiT, H_, 12);
  cvt_tr_kernel<<<(H_ * FF_) / 256, 256, 0, stream>>>(Wo, WoT, FF_, 10);

  // 2) QKV projections (f16 outputs for the attention kernel)
  dim3 gqkv(H_ / 128, NT_ / 128);
  gemm_wmma_kernel<<<gqkv, 256, 0, stream>>>(xf16, WqT, bq, nullptr, qf, NT_, H_, H_, 0);
  gemm_wmma_kernel<<<gqkv, 256, 0, stream>>>(xf16, WkT, bk, nullptr, kf, NT_, H_, H_, 0);
  gemm_wmma_kernel<<<gqkv, 256, 0, stream>>>(xf16, WvT, bv, nullptr, vf, NT_, H_, H_, 0);

  // 3) flash attention
  dim3 ga(S_ / 128, NH_, B_);
  flash_attn_kernel<<<ga, 256, 0, stream>>>(qf, kf, vf, attn);

  // 4) h = LN(x + attn)  (also f16 mirror for FFN GEMM)
  add_ln_kernel<<<NT_, 256, 0, stream>>>(attn, x, g1, b1, h1, h1h);

  // 5) FFN
  dim3 gf1(FF_ / 128, NT_ / 128);
  gemm_wmma_kernel<<<gf1, 256, 0, stream>>>(h1h, WiT, bi, nullptr, f1, NT_, FF_, H_, 1);
  dim3 gf2(H_ / 128, NT_ / 128);
  gemm_wmma_kernel<<<gf2, 256, 0, stream>>>(f1, WoT, bo, f2, nullptr, NT_, H_, FF_, 0);

  // 6) out = LN(h + ffn)
  add_ln_kernel<<<NT_, 256, 0, stream>>>(f2, h1, g2, b2, out, nullptr);
}
